// HyperbolicBertClassifier_90993177133157
// MI455X (gfx1250) — matrix operand order repaired and checked by reading, not compiled, not run on hardware
//
#include <hip/hip_runtime.h>
#include <hip/hip_bf16.h>
#include <math.h>

// ---------------------------------------------------------------------------
// Types for CDNA5 WMMA (wave32): v_wmma_f32_16x16x32_bf16
// ---------------------------------------------------------------------------
typedef __bf16 bf16;
typedef bf16  v16bf __attribute__((ext_vector_type(16)));
typedef bf16  v8bf  __attribute__((ext_vector_type(8)));
typedef float v8f   __attribute__((ext_vector_type(8)));
typedef unsigned int u32x4 __attribute__((ext_vector_type(4)));
typedef unsigned int u32x8 __attribute__((ext_vector_type(8)));

static constexpr int Bn = 32768;  // batch
static constexpr int Hn = 768;    // hidden
static constexpr int Dn = 384;    // ball dim
static constexpr int Ln = 128;    // classes
static constexpr int LDA = 40;    // padded LDS stride (elements) for K=32 tiles

// Native gfx1250 f32 -> bf16 convert (v_cvt class, RNE)
__device__ __forceinline__ bf16 cvt_bf16(float f) { return (bf16)f; }

__device__ __forceinline__ v8f zero8() {
  v8f z;
#pragma unroll
  for (int i = 0; i < 8; ++i) z[i] = 0.0f;
  return z;
}

// Build a 16-element bf16 fragment from two 16-byte LDS chunks (ds_load_b128 x2)
__device__ __forceinline__ v16bf load_frag(const bf16* p0, const bf16* p1) {
  v8bf lo = *(const v8bf*)p0;
  v8bf hi = *(const v8bf*)p1;
  v16bf r;
#pragma unroll
  for (int i = 0; i < 8; ++i) { r[i] = lo[i]; r[i + 8] = hi[i]; }
  return r;
}

// Issue one Tensor Data Mover load (ISA ch.8 D#). Wave-uniform operands in
// SGPR tuples: group0 = 4 SGPRs, group1 = 8 SGPRs. Tracked by TENSORcnt.
__device__ __forceinline__ void tdm_load(unsigned lds_addr,
                                         unsigned long long gaddr,
                                         u32x8 g1) {
  u32x4 g0;
  g0[0] = 1u;                                    // count=1 valid descriptor
  g0[1] = lds_addr;                              // LDS byte offset
  g0[2] = (unsigned)gaddr;                       // global addr lo
  g0[3] = (unsigned)(gaddr >> 32) | 0x80000000u; // addr hi | type=2 (127:126)
  asm volatile("tensor_load_to_lds %0, %1" :: "s"(g0), "s"(g1) : "memory");
}

// ---------------------------------------------------------------------------
// K1: projected = cls @ W + b   (32768x768x384), fused expmap0.
// Block: 64 rows x 384 cols (full D so the row norm is block-local).
// 8 waves: 4 (M) x 2 (N); wave tile 16 x 192 = 12 WMMA accumulators.
// Double-buffered: global loads for tile i+1 issued before tile i's WMMAs.
// ---------------------------------------------------------------------------
__global__ __launch_bounds__(256) void k1_proj_expmap(
    const float* __restrict__ cls, const float* __restrict__ W,
    const float* __restrict__ bias, float* __restrict__ x_out,
    bf16* __restrict__ x_bf, float* __restrict__ x2_out)
{
  __shared__ bf16  As[2][64 * LDA];    // 64 x 32 (bf16, padded)
  __shared__ bf16  Bs[2][384 * LDA];   // N-major: 384 x 32 (bf16, padded)
  __shared__ float rowsum[64];

  const int tid  = threadIdx.x;
  const int lane = tid & 31;
  const int wid  = tid >> 5;
  const int wm   = wid & 3;         // 0..3 : M sub-tile
  const int wn   = wid >> 2;        // 0..1 : N sub-tile
  const int half = lane >> 4;
  const int lcol = lane & 15;
  const int mBlock = blockIdx.x * 64;

  v8f acc[12];
#pragma unroll
  for (int t = 0; t < 12; ++t) acc[t] = zero8();

  float aReg[8], bReg[48];

  // ---- prologue: load+stage tile 0 into buffer 0 ----
#pragma unroll
  for (int i = 0; i < 8; ++i) {
    int idx = tid * 8 + i;
    int r = idx >> 5, c = idx & 31;
    aReg[i] = cls[(size_t)(mBlock + r) * Hn + c];
  }
#pragma unroll
  for (int i = 0; i < 48; ++i) {
    int idx = tid + i * 256;
    int k = idx / 384, n = idx - k * 384;
    bReg[i] = W[(size_t)k * Dn + n];
  }
#pragma unroll
  for (int i = 0; i < 8; ++i) {
    int idx = tid * 8 + i;
    int r = idx >> 5, c = idx & 31;
    As[0][r * LDA + c] = cvt_bf16(aReg[i]);
  }
#pragma unroll
  for (int i = 0; i < 48; ++i) {
    int idx = tid + i * 256;
    int k = idx / 384, n = idx - k * 384;
    Bs[0][n * LDA + k] = cvt_bf16(bReg[i]);
  }
  __syncthreads();

  constexpr int NIT = Hn / 32;   // 24
  for (int it = 0; it < NIT; ++it) {
    const int cur = it & 1;
    // ---- issue global loads for tile it+1 (hide under WMMAs below) ----
    if (it + 1 < NIT) {
      const int kn = (it + 1) * 32;
#pragma unroll
      for (int i = 0; i < 8; ++i) {
        int idx = tid * 8 + i;
        int r = idx >> 5, c = idx & 31;
        aReg[i] = cls[(size_t)(mBlock + r) * Hn + kn + c];
      }
#pragma unroll
      for (int i = 0; i < 48; ++i) {
        int idx = tid + i * 256;
        int k = idx / 384, n = idx - k * 384;
        bReg[i] = W[(size_t)(kn + k) * Dn + n];
      }
    }

    // ---- WMMAs on tile it ----
    const bf16* ap0 = &As[cur][(wm * 16 + lcol) * LDA + half * 8];
    v16bf afrag = load_frag(ap0, ap0 + 16);
#pragma unroll
    for (int t = 0; t < 12; ++t) {
      const bf16* bp0 = &Bs[cur][(wn * 192 + t * 16 + lcol) * LDA + half * 8];
      v16bf bfrag = load_frag(bp0, bp0 + 16);
      acc[t] = __builtin_amdgcn_wmma_f32_16x16x32_bf16(
          false, afrag, false, bfrag, (short)0, acc[t], false, false);
    }

    // ---- convert + store tile it+1 into the idle buffer ----
    if (it + 1 < NIT) {
      const int nxt = cur ^ 1;
#pragma unroll
      for (int i = 0; i < 8; ++i) {
        int idx = tid * 8 + i;
        int r = idx >> 5, c = idx & 31;
        As[nxt][r * LDA + c] = cvt_bf16(aReg[i]);
      }
#pragma unroll
      for (int i = 0; i < 48; ++i) {
        int idx = tid + i * 256;
        int k = idx / 384, n = idx - k * 384;
        Bs[nxt][n * LDA + k] = cvt_bf16(bReg[i]);
      }
      __syncthreads();
    }
  }

  // bias
#pragma unroll
  for (int t = 0; t < 12; ++t) {
    float bv = bias[wn * 192 + t * 16 + lcol];
#pragma unroll
    for (int r = 0; r < 8; ++r) acc[t][r] += bv;
  }

  // per-row sum of squares over this wave's 192 columns, reduce across 16 lanes
  float ss[8];
#pragma unroll
  for (int r = 0; r < 8; ++r) {
    float s = 0.0f;
#pragma unroll
    for (int t = 0; t < 12; ++t) s += acc[t][r] * acc[t][r];
#pragma unroll
    for (int m = 1; m < 16; m <<= 1) s += __shfl_xor(s, m, 32);
    ss[r] = s;
  }

  if (tid < 64) rowsum[tid] = 0.0f;
  __syncthreads();
  if (lcol == 0) {
#pragma unroll
    for (int r = 0; r < 8; ++r)
      atomicAdd(&rowsum[wm * 16 + half * 8 + r], ss[r]);  // ds_add_f32
  }
  __syncthreads();

  // expmap0: x = tanh(n)/n * p ;  x2 = tanh(n)^2
#pragma unroll
  for (int r = 0; r < 8; ++r) {
    int rl = wm * 16 + half * 8 + r;
    size_t rowg = (size_t)(mBlock + rl);
    float n2 = rowsum[rl];
    float n  = sqrtf(fmaxf(n2, 1e-30f));
    n = fmaxf(n, 1e-15f);
    float th = tanhf(n);
    float scale = th / n;
    if (wn == 0 && lcol == 0) x2_out[rowg] = th * th;
#pragma unroll
    for (int t = 0; t < 12; ++t) {
      int col = wn * 192 + t * 16 + lcol;
      float xv = acc[t][r] * scale;
      x_out[rowg * Dn + col] = xv;
      x_bf[rowg * Dn + col]  = cvt_bf16(xv);
    }
  }
}

// ---------------------------------------------------------------------------
// K2: prototypes -> bf16 + y2  (128 rows x 384)
// ---------------------------------------------------------------------------
__global__ __launch_bounds__(384) void k2_prep_y(
    const float* __restrict__ y, bf16* __restrict__ y_bf, float* __restrict__ y2)
{
  __shared__ float red[384];
  int row = blockIdx.x;
  int t = threadIdx.x;
  float v = y[(size_t)row * Dn + t];
  y_bf[(size_t)row * Dn + t] = cvt_bf16(v);
  red[t] = v * v;
  __syncthreads();
  if (t < 128) red[t] += red[t + 128] + red[t + 256];
  __syncthreads();
  for (int s = 64; s > 0; s >>= 1) {
    if (t < s) red[t] += red[t + s];
    __syncthreads();
  }
  if (t == 0) y2[row] = red[0];
}

// ---------------------------------------------------------------------------
// K3: xy = x @ y^T (32768x384x128) fused Poincare distance epilogue.
// Block: 128 x 128. 8 waves: 4 (M) x 2 (N); wave tile 32 x 64 = 2x4 accums.
// Double-buffered Tensor Data Mover pipeline: wave 0 issues the DMA for tile
// i+1, then s_wait_tensorcnt 2 (in-order completion => tile i done) so the
// next tile's DMA overlaps this tile's WMMAs. LDS row pad via D# pad fields:
// pad_interval=3 (16 DW = 64B row), pad_amount=3 (4 DW) -> 80B == LDA bf16.
// ---------------------------------------------------------------------------
__global__ __launch_bounds__(256) void k3_dist(
    const bf16* __restrict__ x_bf, const bf16* __restrict__ y_bf,
    const float* __restrict__ x2, const float* __restrict__ y2,
    float* __restrict__ logits)
{
  __shared__ bf16 As[2][128 * LDA];
  __shared__ bf16 Bs[2][128 * LDA];

  const int tid  = threadIdx.x;
  const int lane = tid & 31;
  const int wid  = tid >> 5;
  const int wm   = wid & 3;
  const int wn   = wid >> 2;
  const int half = lane >> 4;
  const int lcol = lane & 15;
  const int mBlock = blockIdx.x * 128;

  v8f acc[2][4];
#pragma unroll
  for (int mt = 0; mt < 2; ++mt)
#pragma unroll
    for (int nt = 0; nt < 4; ++nt) acc[mt][nt] = zero8();

  // TDM descriptor group 1 (shared by both tiles)
  u32x8 g1;
  g1[0] = (1u << 16)        // data_size = 2 bytes
        | (1u << 20)        // pad_enable
        | (3u << 22)        // pad_interval: 16 DWORDs (one 64B row)
        | (3u << 25);       // pad_amount:   4 DWORDs (16B)  -> stride 80B
  g1[1] = (384u & 0xFFFFu) << 16;              // tensor_dim0 = 384 (lo bits)
  g1[2] = (384u >> 16) | (32768u << 16);       // tensor_dim0 hi | tensor_dim1 lo
  g1[3] = (32u << 16);                         // tile_dim0 = 32 (k)
  g1[4] = 128u;                                // tile_dim1 = 128 (rows)
  g1[5] = 384u;                                // tensor_dim0_stride = 384
  g1[6] = 0u;
  g1[7] = 0u;

  const unsigned ldsA[2] = {(unsigned)(uintptr_t)&As[0][0],
                            (unsigned)(uintptr_t)&As[1][0]};
  const unsigned ldsB[2] = {(unsigned)(uintptr_t)&Bs[0][0],
                            (unsigned)(uintptr_t)&Bs[1][0]};
  const unsigned long long baseA =
      (unsigned long long)(uintptr_t)x_bf + (size_t)mBlock * Dn * 2ull;
  const unsigned long long baseB = (unsigned long long)(uintptr_t)y_bf;

  constexpr int NIT = Dn / 32;   // 12
  if (wid == 0) {
    tdm_load(ldsA[0], baseA, g1);
    tdm_load(ldsB[0], baseB, g1);
  }

#pragma unroll
  for (int it = 0; it < NIT; ++it) {
    const int cur = it & 1;
    __syncthreads();   // all waves done reading buffer cur^1 (iteration it-1)
    if (wid == 0) {
      if (it + 1 < NIT) {
        const unsigned long long koff = (unsigned long long)(it + 1) * 64ull;
        tdm_load(ldsA[cur ^ 1], baseA + koff, g1);
        tdm_load(ldsB[cur ^ 1], baseB + koff, g1);
        __builtin_amdgcn_s_wait_tensorcnt(2);  // oldest 2 (tile it) complete
      } else {
        __builtin_amdgcn_s_wait_tensorcnt(0);
      }
    }
    __syncthreads();   // tile it visible to all waves

#pragma unroll
    for (int mt = 0; mt < 2; ++mt) {
      const bf16* ap0 = &As[cur][(wm * 32 + mt * 16 + lcol) * LDA + half * 8];
      v16bf a = load_frag(ap0, ap0 + 16);
#pragma unroll
      for (int nt = 0; nt < 4; ++nt) {
        const bf16* bp0 = &Bs[cur][(wn * 64 + nt * 16 + lcol) * LDA + half * 8];
        v16bf b = load_frag(bp0, bp0 + 16);
        acc[mt][nt] = __builtin_amdgcn_wmma_f32_16x16x32_bf16(
            false, a, false, b, (short)0, acc[mt][nt], false, false);
      }
    }
  }

  // Poincare distance epilogue:
  // ratio = clip((x2+y2-2xy)/(1-2xy+x2*y2), 0, (1-1e-5)^2)
  // logits = -2*atanh(sqrt(ratio)) = log((1-s)/(1+s))
  const float RMAX = 0.99998000009999f;
  float y2v[4];
#pragma unroll
  for (int nt = 0; nt < 4; ++nt) y2v[nt] = y2[wn * 64 + nt * 16 + lcol];

#pragma unroll
  for (int mt = 0; mt < 2; ++mt) {
#pragma unroll
    for (int r = 0; r < 8; ++r) {
      size_t rowg = (size_t)(mBlock + wm * 32 + mt * 16 + half * 8 + r);
      float xx = x2[rowg];
#pragma unroll
      for (int nt = 0; nt < 4; ++nt) {
        float xy  = acc[mt][nt][r];
        float sq  = xx + y2v[nt] - 2.0f * xy;
        float den = 1.0f - 2.0f * xy + xx * y2v[nt];
        float ratio = fminf(fmaxf(sq / den, 0.0f), RMAX);
        float s = sqrtf(ratio);
        float lg = __logf((1.0f - s) / (1.0f + s));
        logits[rowg * Ln + wn * 64 + nt * 16 + lcol] = lg;
      }
    }
  }
}

// ---------------------------------------------------------------------------
extern "C" void kernel_launch(void* const* d_in, const int* in_sizes, int n_in,
                              void* d_out, int out_size, void* d_ws, size_t ws_size,
                              hipStream_t stream) {
  (void)in_sizes; (void)n_in; (void)out_size; (void)ws_size;
  const float* cls  = (const float*)d_in[0];   // [B,H]
  const float* W    = (const float*)d_in[1];   // [H,D]
  const float* bias = (const float*)d_in[2];   // [D]
  const float* pw   = (const float*)d_in[3];   // [L,D]

  float* logits = (float*)d_out;                         // [B,L]
  float* x_out  = (float*)d_out + (size_t)Bn * Ln;       // [B,D]

  char* ws = (char*)d_ws;
  bf16* x_bf = (bf16*)ws;
  size_t off = (size_t)Bn * Dn * sizeof(bf16);           // 25.2 MB
  float* x2 = (float*)(ws + off); off += (size_t)Bn * sizeof(float);
  bf16* y_bf = (bf16*)(ws + off); off += (size_t)Ln * Dn * sizeof(bf16);
  float* y2 = (float*)(ws + off);

  hipLaunchKernelGGL(k1_proj_expmap, dim3(Bn / 64), dim3(256), 0, stream,
                     cls, W, bias, x_out, x_bf, x2);
  hipLaunchKernelGGL(k2_prep_y, dim3(Ln), dim3(384), 0, stream, pw, y_bf, y2);
  hipLaunchKernelGGL(k3_dist, dim3(Bn / 128), dim3(256), 0, stream,
                     x_bf, y_bf, x2, y2, logits);
}
